// Encoder_19353122635841
// MI455X (gfx1250) — compile-verified, matrix operand order checked by reading, not run on hardware
//
#include <hip/hip_runtime.h>
#include <hip/hip_bf16.h>

// ---------------------------------------------------------------------------
// Transformer encoder layer for MI455X (gfx1250, wave32, WMMA).
// All GEMMs (QKV, QK^T, PV, proj, FFN1, FFN2) run on v_wmma_f32_16x16x32_f16.
// GEMM A-operands are staged global->LDS with async-tensor copies
// (global_load_async_to_lds_b128 / s_wait_asynccnt), double-buffered.
// ---------------------------------------------------------------------------

#define D_     584          // embed dim
#define DP_    608          // D padded to 19*32
#define H_     8            // heads
#define HD_    73           // head dim
#define HDP_   96           // head dim padded to 3*32
#define DFF_   2336         // 4*D = 73*32 (already WMMA-friendly)
#define T_     2048
#define B_     4
#define ROWS_  8192         // B*T
#define SCALE_ 0.0413802938f // 584^-0.5  (reference scales by D^-0.5)

typedef _Float16 half8 __attribute__((ext_vector_type(8)));
typedef _Float16 v16h  __attribute__((ext_vector_type(16)));
typedef float    v8f   __attribute__((ext_vector_type(8)));

__device__ __forceinline__ v8f wmma_f16(v16h a, v16h b, v8f c) {
  // D = A(16x32,f16) * B(32x16,f16) + C(16x16,f32)
  return __builtin_amdgcn_wmma_f32_16x16x32_f16(false, a, false, b, (short)0, c,
                                                false, false);
}

// Load a 16-bit A-matrix fragment (ISA 7.12.2 layout) from a row-major row:
// lane<16 holds K {0..7,16..23}, lane>=16 holds K {8..15,24..31}.
// `base` points at row_start + kchunk*32 (16B aligned), caller passes lane.
__device__ __forceinline__ v16h load_afrag(const _Float16* base, int lane) {
  const _Float16* p = base + ((lane & 16) ? 8 : 0);
  half8 lo = *(const half8*)(p);
  half8 hi = *(const half8*)(p + 16);
  return __builtin_shufflevector(lo, hi, 0, 1, 2, 3, 4, 5, 6, 7,
                                 8, 9, 10, 11, 12, 13, 14, 15);
}

// Generic-pointer low 32 bits == wave-relative LDS byte offset (aperture rule).
__device__ __forceinline__ unsigned lds_off(const void* p) {
  return (unsigned)(size_t)p;
}

// ---------------------------------------------------------------------------
// Weight pre-swizzle: fp32 [nmat][Kreal][Nreal] row-major -> f16 B-fragment
// order [mat][kt][nt][lane][16], zero-padded, so GEMM B loads are one
// contiguous 32B v16h per lane.
// ---------------------------------------------------------------------------
__global__ __launch_bounds__(256) void pack_b(const float* __restrict__ src,
                                              _Float16* __restrict__ dst,
                                              int Kreal, int Nreal, int KT,
                                              int NTp, int nmat) {
  long idx = (long)blockIdx.x * 256 + threadIdx.x;
  long per = (long)KT * NTp * 512;
  if (idx >= per * nmat) return;
  int mat   = (int)(idx / per);
  long r    = idx - (long)mat * per;
  int j     = (int)(r & 15);
  int lane  = (int)((r >> 4) & 31);
  long tile = r >> 9;
  int nt = (int)(tile % NTp);
  int kt = (int)(tile / NTp);
  int n = nt * 16 + (lane & 15);
  int k = kt * 32 + ((j < 8) ? j : j + 8) + ((lane & 16) ? 8 : 0);
  float v = 0.f;
  if (k < Kreal && n < Nreal)
    v = src[(size_t)mat * Kreal * Nreal + (size_t)k * Nreal + n];
  dst[idx] = (_Float16)v;
}

__global__ __launch_bounds__(256) void zero_f16(_Float16* __restrict__ p, long n) {
  long i = (long)blockIdx.x * 256 + threadIdx.x;
  if (i < n) p[i] = (_Float16)0.f;
}

// ---------------------------------------------------------------------------
// LayerNorm: fp32 [row][584] -> f16 [row][608] (pad cols zeroed).
// ---------------------------------------------------------------------------
__global__ __launch_bounds__(256) void ln_kernel(const float* __restrict__ src,
                                                 const float* __restrict__ g,
                                                 const float* __restrict__ be,
                                                 _Float16* __restrict__ dst) {
  const int row = blockIdx.x;
  const float* xr = src + (size_t)row * D_;
  __shared__ float red[256];
  float s = 0.f;
  for (int i = threadIdx.x; i < D_; i += 256) s += xr[i];
  red[threadIdx.x] = s;
  __syncthreads();
  for (int st = 128; st > 0; st >>= 1) {
    if (threadIdx.x < st) red[threadIdx.x] += red[threadIdx.x + st];
    __syncthreads();
  }
  float mu = red[0] / (float)D_;
  __syncthreads();
  float v = 0.f;
  for (int i = threadIdx.x; i < D_; i += 256) {
    float d = xr[i] - mu;
    v += d * d;
  }
  red[threadIdx.x] = v;
  __syncthreads();
  for (int st = 128; st > 0; st >>= 1) {
    if (threadIdx.x < st) red[threadIdx.x] += red[threadIdx.x + st];
    __syncthreads();
  }
  float rstd = rsqrtf(red[0] / (float)D_ + 1e-5f);
  for (int i = threadIdx.x; i < DP_; i += 256) {
    float val = 0.f;
    if (i < D_) val = (xr[i] - mu) * rstd * g[i] + be[i];
    dst[(size_t)row * DP_ + i] = (_Float16)val;
  }
}

// ---------------------------------------------------------------------------
// Generic WMMA GEMM: C[8192 x N] = A[8192 x K] * B[K x N], f16 in, f32 acc.
// 4 waves per block (2x2); block tile 64x64; 2x2 frags per wave.
// A tile staged global->LDS via async copies in groups of 4 K-chunks
// (64x128 halfs = 16KB), double-buffered; prefetch of group g+1 overlaps
// WMMA on group g. B operands load directly from fragment-order weights.
// ---------------------------------------------------------------------------
enum { EPI_QK = 0, EPI_V = 1, EPI_BR = 2, EPI_RELU = 3 };

template <int EPI>
__global__ __launch_bounds__(128) void gemm_ws(
    const _Float16* __restrict__ A, int lda, const _Float16* __restrict__ Bp,
    int KT, int NTp, _Float16* __restrict__ hout, float* __restrict__ fout,
    const float* __restrict__ bias, const float* __restrict__ resid, int ncol) {
  __shared__ __align__(16) _Float16 sA[2][64 * 128];  // 2 x 16KB

  const int tid = threadIdx.x;
  const int lane = tid & 31;
  const int wave = tid >> 5;
  const int wm = wave >> 1, wn = wave & 1;
  const int rowBase = blockIdx.x * 64 + wm * 32;
  const int ct0 = blockIdx.y * 4 + wn * 2;   // first 16-col tile for this wave
  const int lsel = lane & 15;
  const _Float16* Bm = Bp + (size_t)blockIdx.z * ((size_t)KT * NTp * 512);

  // async staging: thread t copies 32B per chunk of row (t>>1), seg (t&1)
  const int arow = tid >> 1;
  const int aseg = (tid & 1) * 16;  // halfs
  const _Float16* gRow = A + (size_t)(blockIdx.x * 64 + arow) * lda + aseg;

  auto stage = [&](int g, int buf) {
    unsigned lbase = lds_off(&sA[buf][arow * 128 + aseg]);
#pragma unroll
    for (int c = 0; c < 4; ++c) {
      const int kt = g * 4 + c;
      if (kt >= KT) break;
      const _Float16* gp = gRow + kt * 32;
      unsigned lp = lbase + c * 64;  // c*32 halfs
      asm volatile(
          "global_load_async_to_lds_b128 %0, %1, off\n\t"
          "global_load_async_to_lds_b128 %0, %1, off offset:16"
          :
          : "v"(lp), "v"(gp)
          : "memory");
    }
  };

  v8f acc[2][2] = {};
  const int G = (KT + 3) >> 2;
  stage(0, 0);
  asm volatile("s_wait_asynccnt 0x0" ::: "memory");
  __syncthreads();

  for (int g = 0; g < G; ++g) {
    const int cur = g & 1;
    if (g + 1 < G) stage(g + 1, cur ^ 1);
    const _Float16* sR0 = &sA[cur][(wm * 32 + lsel) * 128];
    const _Float16* sR1 = &sA[cur][(wm * 32 + 16 + lsel) * 128];
#pragma unroll
    for (int c = 0; c < 4; ++c) {
      const int kt = g * 4 + c;
      if (kt >= KT) break;
      v16h af0 = load_afrag(sR0 + c * 32, lane);
      v16h af1 = load_afrag(sR1 + c * 32, lane);
      v16h bf0 = *(const v16h*)(Bm + (((size_t)kt * NTp + ct0) * 32 + lane) * 16);
      v16h bf1 =
          *(const v16h*)(Bm + (((size_t)kt * NTp + ct0 + 1) * 32 + lane) * 16);
      acc[0][0] = wmma_f16(af0, bf0, acc[0][0]);
      acc[0][1] = wmma_f16(af0, bf1, acc[0][1]);
      acc[1][0] = wmma_f16(af1, bf0, acc[1][0]);
      acc[1][1] = wmma_f16(af1, bf1, acc[1][1]);
    }
    asm volatile("s_wait_asynccnt 0x0" ::: "memory");
    __syncthreads();
  }

#pragma unroll
  for (int mt = 0; mt < 2; ++mt)
#pragma unroll
    for (int nt = 0; nt < 2; ++nt) {
      const int col = (ct0 + nt) * 16 + lsel;
      if (col >= ncol) continue;
#pragma unroll
      for (int i = 0; i < 8; ++i) {
        const int row = rowBase + mt * 16 + ((lane & 16) ? 8 + i : i);
        float v = acc[mt][nt][i];
        if constexpr (EPI == EPI_QK) {
          hout[((size_t)blockIdx.z * ROWS_ + row) * HDP_ + col] = (_Float16)v;
        } else if constexpr (EPI == EPI_V) {
          int b = row >> 11, t = row & 2047;
          hout[((size_t)(blockIdx.z * B_ + b) * HDP_ + col) * T_ + t] =
              (_Float16)v;
        } else if constexpr (EPI == EPI_BR) {
          size_t o = (size_t)row * D_ + col;
          fout[o] = v + bias[col] + resid[o];
        } else {  // EPI_RELU
          hout[(size_t)row * DFF_ + col] = (_Float16)fmaxf(v + bias[col], 0.f);
        }
      }
    }
}

// ---------------------------------------------------------------------------
// Flash attention: per (b, h, 64-row q tile). WMMA QK^T -> LDS -> online
// softmax by 64 row-threads -> WMMA P*V with running-max rescale of O.
// ---------------------------------------------------------------------------
__global__ __launch_bounds__(128) void attn_kernel(
    const _Float16* __restrict__ Q, const _Float16* __restrict__ Kmat,
    const _Float16* __restrict__ VT, _Float16* __restrict__ attn) {
  const int b = blockIdx.z, h = blockIdx.y;
  const int qbase = blockIdx.x * 64;
  const int tid = threadIdx.x, lane = tid & 31, wave = tid >> 5;
  const int wm = wave >> 1, wn = wave & 1;
  const int lsel = lane & 15;
  const int hsel = (lane & 16) ? 8 : 0;

  __shared__ __align__(16) float sS[64][68];
  __shared__ __align__(16) _Float16 sP[64][64];
  __shared__ float sC[64];
  __shared__ float sL[64];

  const _Float16* Qh = Q + ((size_t)h * ROWS_ + (size_t)b * T_) * HDP_;
  const _Float16* Kp = Kmat + ((size_t)h * ROWS_ + (size_t)b * T_) * HDP_;
  const _Float16* Vp = VT + ((size_t)(h * B_ + b) * HDP_) * T_;

  v16h qf[2][3];
#pragma unroll
  for (int mt = 0; mt < 2; ++mt)
#pragma unroll
    for (int kt = 0; kt < 3; ++kt)
      qf[mt][kt] = load_afrag(
          Qh + (size_t)(qbase + wm * 32 + mt * 16 + lsel) * HDP_ + kt * 32, lane);

  v8f of[2][3] = {};               // O accum: rows wm*32+, cols wn*48+
  float m_r = -1e30f, l_r = 0.f;   // running max/sum, only tid<64

  for (int sbase = 0; sbase < T_; sbase += 64) {
    // ---- scores quadrant: S = Q * K^T (B-frag == A-pattern load of K rows)
    v16h kf[2][3];
#pragma unroll
    for (int nt = 0; nt < 2; ++nt)
#pragma unroll
      for (int kt = 0; kt < 3; ++kt)
        kf[nt][kt] = load_afrag(
            Kp + (size_t)(sbase + wn * 32 + nt * 16 + lsel) * HDP_ + kt * 32,
            lane);
    v8f sc[2][2] = {};
#pragma unroll
    for (int mt = 0; mt < 2; ++mt)
#pragma unroll
      for (int nt = 0; nt < 2; ++nt)
#pragma unroll
        for (int kt = 0; kt < 3; ++kt)
          sc[mt][nt] = wmma_f16(qf[mt][kt], kf[nt][kt], sc[mt][nt]);
#pragma unroll
    for (int mt = 0; mt < 2; ++mt)
#pragma unroll
      for (int nt = 0; nt < 2; ++nt)
#pragma unroll
        for (int i = 0; i < 8; ++i)
          sS[wm * 32 + mt * 16 + hsel + i][wn * 32 + nt * 16 + lsel] =
              sc[mt][nt][i] * SCALE_;
    __syncthreads();

    // ---- online softmax: one thread per row
    if (tid < 64) {
      float mx = -1e30f;
      for (int j = 0; j < 64; ++j) mx = fmaxf(mx, sS[tid][j]);
      float mnew = fmaxf(m_r, mx);
      float c = __expf(m_r - mnew);
      float ssum = 0.f;
      for (int j = 0; j < 64; ++j) {
        float p = __expf(sS[tid][j] - mnew);
        ssum += p;
        sP[tid][j] = (_Float16)p;
      }
      l_r = l_r * c + ssum;
      m_r = mnew;
      sC[tid] = c;
    }
    __syncthreads();

    // ---- O = O*c + P*V
    v16h pf[2][2], vf[3][2];
#pragma unroll
    for (int mt = 0; mt < 2; ++mt)
#pragma unroll
      for (int k2 = 0; k2 < 2; ++k2)
        pf[mt][k2] = load_afrag(&sP[wm * 32 + mt * 16 + lsel][k2 * 32], lane);
#pragma unroll
    for (int ot = 0; ot < 3; ++ot)
#pragma unroll
      for (int k2 = 0; k2 < 2; ++k2)
        vf[ot][k2] = load_afrag(
            Vp + (size_t)(wn * 48 + ot * 16 + lsel) * T_ + sbase + k2 * 32, lane);
#pragma unroll
    for (int mt = 0; mt < 2; ++mt)
#pragma unroll
      for (int ot = 0; ot < 3; ++ot) {
#pragma unroll
        for (int i = 0; i < 8; ++i)
          of[mt][ot][i] *= sC[wm * 32 + mt * 16 + hsel + i];
#pragma unroll
        for (int k2 = 0; k2 < 2; ++k2)
          of[mt][ot] = wmma_f16(pf[mt][k2], vf[ot][k2], of[mt][ot]);
      }
    __syncthreads();
  }

  if (tid < 64) sL[tid] = 1.f / l_r;
  __syncthreads();
#pragma unroll
  for (int mt = 0; mt < 2; ++mt)
#pragma unroll
    for (int ot = 0; ot < 3; ++ot) {
      const int j = wn * 48 + ot * 16 + lsel;
      if (j >= HD_) continue;  // drop padded head cols
#pragma unroll
      for (int i = 0; i < 8; ++i) {
        const int rr = wm * 32 + mt * 16 + hsel + i;
        attn[((size_t)(b * T_ + qbase + rr)) * DP_ + h * HD_ + j] =
            (_Float16)(of[mt][ot][i] * sL[rr]);
      }
    }
}

// ---------------------------------------------------------------------------
extern "C" void kernel_launch(void* const* d_in, const int* in_sizes, int n_in,
                              void* d_out, int out_size, void* d_ws,
                              size_t ws_size, hipStream_t stream) {
  (void)in_sizes; (void)n_in; (void)out_size; (void)ws_size;
  const float* x     = (const float*)d_in[0];
  const float* Wq    = (const float*)d_in[1];
  const float* Wk    = (const float*)d_in[2];
  const float* Wv    = (const float*)d_in[3];
  const float* Wproj = (const float*)d_in[4];
  const float* bproj = (const float*)d_in[5];
  const float* W1    = (const float*)d_in[6];
  const float* b1    = (const float*)d_in[7];
  const float* W2    = (const float*)d_in[8];
  const float* b2    = (const float*)d_in[9];
  const float* g1    = (const float*)d_in[10];
  const float* be1   = (const float*)d_in[11];
  const float* g2    = (const float*)d_in[12];
  const float* be2   = (const float*)d_in[13];
  float* out = (float*)d_out;

  char* ws = (char*)d_ws;
  size_t off = 0;
  auto take = [&](size_t bytes) -> char* {
    char* p = ws + off;
    off = (off + bytes + 255) & ~(size_t)255;
    return p;
  };

  // packed f16 weights (B-fragment order)
  const size_t qkvPack = (size_t)H_ * 19 * 8 * 512;   // per-matrix halfs
  _Float16* pWq = (_Float16*)take(qkvPack * 2);
  _Float16* pWk = (_Float16*)take(qkvPack * 2);
  _Float16* pWv = (_Float16*)take(qkvPack * 2);
  _Float16* pWp = (_Float16*)take((size_t)19 * 40 * 512 * 2);
  _Float16* pW1 = (_Float16*)take((size_t)19 * 148 * 512 * 2);
  _Float16* pW2 = (_Float16*)take((size_t)73 * 40 * 512 * 2);
  // activations
  _Float16* xn   = (_Float16*)take((size_t)ROWS_ * DP_ * 2);   // ln1, reused ln2
  char*     qoff = take((size_t)H_ * ROWS_ * HDP_ * 2);
  _Float16* qb   = (_Float16*)qoff;
  _Float16* kb   = (_Float16*)take((size_t)H_ * ROWS_ * HDP_ * 2);
  _Float16* vtb  = (_Float16*)take((size_t)H_ * ROWS_ * HDP_ * 2);
  _Float16* attnb = (_Float16*)take((size_t)ROWS_ * DP_ * 2);
  float*    x1   = (float*)take((size_t)ROWS_ * D_ * 4);
  // FFN hidden overlays the (dead by then) q/k/vT/attn region
  _Float16* hb = (_Float16*)qoff;  // 8192*2336*2 = 38.3MB < 47.7MB region

  // ---- 1. pack weights to f16 fragment order
  {
    long t;
    t = (long)qkvPack;
    pack_b<<<dim3((t + 255) / 256), 256, 0, stream>>>(Wq, pWq, D_, HD_, 19, 8, H_);
    pack_b<<<dim3((t + 255) / 256), 256, 0, stream>>>(Wk, pWk, D_, HD_, 19, 8, H_);
    pack_b<<<dim3((t + 255) / 256), 256, 0, stream>>>(Wv, pWv, D_, HD_, 19, 8, H_);
    t = (long)19 * 40 * 512;
    pack_b<<<dim3((t + 255) / 256), 256, 0, stream>>>(Wproj, pWp, D_, D_, 19, 40, 1);
    t = (long)19 * 148 * 512;
    pack_b<<<dim3((t + 255) / 256), 256, 0, stream>>>(W1, pW1, D_, DFF_, 19, 148, 1);
    t = (long)73 * 40 * 512;
    pack_b<<<dim3((t + 255) / 256), 256, 0, stream>>>(W2, pW2, DFF_, D_, 73, 40, 1);
  }
  // zero attn buffer (pad cols 584..607 are never written by attention)
  {
    long n = (long)ROWS_ * DP_;
    zero_f16<<<dim3((n + 255) / 256), 256, 0, stream>>>(attnb, n);
  }

  // ---- 2. LN1 -> xn (f16, padded)
  ln_kernel<<<dim3(ROWS_), 256, 0, stream>>>(x, g1, be1, xn);

  // ---- 3. QKV projections (per head, N padded to 128)
  gemm_ws<EPI_QK><<<dim3(ROWS_ / 64, 2, H_), 128, 0, stream>>>(
      xn, DP_, pWq, 19, 8, qb, nullptr, nullptr, nullptr, HDP_);
  gemm_ws<EPI_QK><<<dim3(ROWS_ / 64, 2, H_), 128, 0, stream>>>(
      xn, DP_, pWk, 19, 8, kb, nullptr, nullptr, nullptr, HDP_);
  gemm_ws<EPI_V><<<dim3(ROWS_ / 64, 2, H_), 128, 0, stream>>>(
      xn, DP_, pWv, 19, 8, vtb, nullptr, nullptr, nullptr, HDP_);

  // ---- 4. flash attention -> attnb (concat heads, f16)
  attn_kernel<<<dim3(T_ / 64, H_, B_), 128, 0, stream>>>(qb, kb, vtb, attnb);

  // ---- 5. output projection + bias + residual -> x1 (fp32)
  gemm_ws<EPI_BR><<<dim3(ROWS_ / 64, 10, 1), 128, 0, stream>>>(
      attnb, DP_, pWp, 19, 40, nullptr, x1, bproj, x, D_);

  // ---- 6. LN2 -> xn (reuse)
  ln_kernel<<<dim3(ROWS_), 256, 0, stream>>>(x1, g2, be2, xn);

  // ---- 7. FFN1: relu(xn*W1 + b1) -> hb (f16)
  gemm_ws<EPI_RELU><<<dim3(ROWS_ / 64, 37, 1), 128, 0, stream>>>(
      xn, DP_, pW1, 19, 148, hb, nullptr, b1, nullptr, DFF_);

  // ---- 8. FFN2: hb*W2 + b2 + x1 -> out (fp32)
  gemm_ws<EPI_BR><<<dim3(ROWS_ / 64, 10, 1), 128, 0, stream>>>(
      hb, DFF_, pW2, 73, 40, nullptr, out, b2, x1, D_);
}
